// Attention_790273982864
// MI455X (gfx1250) — compile-verified, matrix operand order checked by reading, not run on hardware
//
#include <hip/hip_runtime.h>
#include <hip/hip_bf16.h>
#include <math.h>

// ---------------------------------------------------------------------------
// Fused multi-head attention for MI455X (gfx1250, wave32, WMMA).
//   B=2, N=2048, D=1024, H=16, HD=64
// Pipeline:
//   1) convert X to f16; convert+transpose Wq/Wk/Wv/Wo to f16 [N][K] layout
//   2) QKV projection GEMM (f16 WMMA, fp32 accum, async global->LDS tiles)
//      -> Q [BH,N,64] (pre-scaled by 1/sqrt(64)), K [BH,N,64], V^T [BH,64,N]
//   3) flash-attention per (bh, 64-row q tile): S = Q K^T via WMMA, online
//      softmax in LDS, O += P V via WMMA. Scores never touch HBM.
//   4) output projection GEMM -> fp32 out + bias
// ---------------------------------------------------------------------------

typedef __attribute__((ext_vector_type(16))) _Float16 v16h;
typedef __attribute__((ext_vector_type(8)))  _Float16 v8h;
typedef __attribute__((ext_vector_type(8)))  float    v8f;

#define B_   2
#define N_   2048
#define D_   1024
#define H_   16
#define HD_  64
#define M_   (B_ * N_)      // 4096 total rows

// ---------------------------------------------------------------------------
// CDNA5 async global->LDS copy (16 bytes per lane), tracked by ASYNCcnt.
// VGLOBAL encoding: VDST = LDS byte address, VADDR = 64-bit global address.
// ---------------------------------------------------------------------------
__device__ __forceinline__ void async_g2l_b128(unsigned lds_off,
                                               const void* gptr) {
  asm volatile("global_load_async_to_lds_b128 %0, %1, off"
               :: "v"(lds_off), "v"(gptr)
               : "memory");
}
__device__ __forceinline__ void wait_async0() {
  asm volatile("s_wait_asynccnt 0x0" ::: "memory");
}
__device__ __forceinline__ unsigned lds_addr(const void* p) {
  return (unsigned)(uintptr_t)p;   // low 32 bits of generic ptr = LDS offset
}

// ---------------------------------------------------------------------------
// Fragment loader.
// 16-bit A-fragment (16x32), wave32 layout (ISA 7.12.2):
//   lane L, element e  ->  A[L%16][(e<8 ? e : e+8) + (L<16 ? 0 : 8)]
// i.e. two contiguous 8-half (16-byte) loads at k-offsets {g, 16+g}, g = hi*8.
// B-fragment uses the same pattern with N = L%16, read from an [N][K] buffer.
// ---------------------------------------------------------------------------
__device__ __forceinline__ v16h load_frag(const _Float16* __restrict__ base,
                                          int ld, int lo, int hi) {
  const _Float16* p = base + lo * ld + hi * 8;
  v8h a = *(const v8h*)(p);
  v8h b = *(const v8h*)(p + 16);
  v16h r;
#pragma unroll
  for (int i = 0; i < 8; ++i) { r[i] = a[i]; r[i + 8] = b[i]; }
  return r;
}

__device__ __forceinline__ v8f wmma_f16(v16h a, v16h b, v8f c) {
  return __builtin_amdgcn_wmma_f32_16x16x32_f16(false, a, false, b,
                                                (short)0, c, false, false);
}

// ---------------------------------------------------------------------------
// fp32 -> f16 convert (elementwise, for X)
// ---------------------------------------------------------------------------
__global__ void f32_to_f16(const float* __restrict__ in,
                           _Float16* __restrict__ out, int n) {
  int i = blockIdx.x * blockDim.x + threadIdx.x;
  int stride = gridDim.x * blockDim.x;
  for (; i < n; i += stride) out[i] = (_Float16)in[i];
}

// ---------------------------------------------------------------------------
// fp32 [K][N] -> f16 [N][K] convert + transpose (for weights), 32x32 tiles.
// ---------------------------------------------------------------------------
__global__ __launch_bounds__(256) void convT_f16(const float* __restrict__ in,
                                                 _Float16* __restrict__ outT) {
  __shared__ _Float16 t[32][33];
  const int kb = blockIdx.y * 32;
  const int nb = blockIdx.x * 32;
  const int tx = threadIdx.x & 31;
  const int ty = threadIdx.x >> 5;  // 0..7
#pragma unroll
  for (int i = 0; i < 4; ++i) {
    const int k = ty + i * 8;
    t[k][tx] = (_Float16)in[(size_t)(kb + k) * D_ + nb + tx];
  }
  __syncthreads();
#pragma unroll
  for (int i = 0; i < 4; ++i) {
    const int n = ty + i * 8;
    outT[(size_t)(nb + n) * D_ + kb + tx] = t[tx][n];
  }
}

// ---------------------------------------------------------------------------
// QKV projection GEMM. 256 threads = 8 waves; tile 128(M) x 64(N), K-step 32.
// A tile (Xh, row-major) and B tile (W^T, [n][k]) are both staged with
// global_load_async_to_lds_b128. Waves: wm = w&3 (M sub), wn = w>>2 (N sub);
// each wave holds 2x2 f32 16x16 accumulators. blockIdx.z selects Q / K / V.
// ---------------------------------------------------------------------------
__global__ __launch_bounds__(256) void qkv_gemm(
    const _Float16* __restrict__ Xh,
    const _Float16* __restrict__ WqT, const _Float16* __restrict__ WkT,
    const _Float16* __restrict__ WvT,
    const float* __restrict__ bq, const float* __restrict__ bk,
    const float* __restrict__ bv,
    _Float16* __restrict__ Qh, _Float16* __restrict__ Kh,
    _Float16* __restrict__ Vt) {
  const int mode = blockIdx.z;
  const _Float16* WT  = (mode == 0) ? WqT : (mode == 1) ? WkT : WvT;
  const float*    bia = (mode == 0) ? bq  : (mode == 1) ? bk  : bv;

  const int Mb = blockIdx.y * 128;
  const int Nb = blockIdx.x * 64;
  const int tid = threadIdx.x;
  const int w = tid >> 5, lane = tid & 31, lo = lane & 15, hi = lane >> 4;
  const int wm = w & 3, wn = w >> 2;

  __shared__ __align__(16) _Float16 As[128][40];  // 128(M) x 32(K), padded
  __shared__ __align__(16) _Float16 Bs[64][40];   // 64(N)  x 32(K), padded

  v8f acc[2][2] = {};

  // per-thread tile-staging coordinates (constant across K loop)
  const int ar = tid >> 1;            // A row 0..127
  const int ac = (tid & 1) * 16;      // A col 0 / 16
  const int bn = tid >> 2;            // B row (n) 0..63
  const int bk8 = (tid & 3) * 8;      // B col (k) 0/8/16/24
  const unsigned a_dst = lds_addr(&As[ar][ac]);
  const unsigned b_dst = lds_addr(&Bs[bn][bk8]);

  for (int k0 = 0; k0 < D_; k0 += 32) {
    __syncthreads();  // previous iteration's fragment reads complete
    {
      const _Float16* srcA = Xh + (size_t)(Mb + ar) * D_ + k0 + ac;
      async_g2l_b128(a_dst,      srcA);
      async_g2l_b128(a_dst + 16, srcA + 8);
      const _Float16* srcB = WT + (size_t)(Nb + bn) * D_ + k0 + bk8;
      async_g2l_b128(b_dst, srcB);
    }
    wait_async0();
    __syncthreads();

    v16h af[2], bf[2];
#pragma unroll
    for (int mi = 0; mi < 2; ++mi)
      af[mi] = load_frag(&As[wm * 32 + mi * 16][0], 40, lo, hi);
#pragma unroll
    for (int ni = 0; ni < 2; ++ni)
      bf[ni] = load_frag(&Bs[wn * 32 + ni * 16][0], 40, lo, hi);
#pragma unroll
    for (int mi = 0; mi < 2; ++mi)
#pragma unroll
      for (int ni = 0; ni < 2; ++ni)
        acc[mi][ni] = wmma_f16(af[mi], bf[ni], acc[mi][ni]);
  }

#pragma unroll
  for (int mi = 0; mi < 2; ++mi)
#pragma unroll
    for (int ni = 0; ni < 2; ++ni)
#pragma unroll
      for (int r = 0; r < 8; ++r) {
        const int m = Mb + wm * 32 + mi * 16 + hi * 8 + r;
        const int c = Nb + wn * 32 + ni * 16 + lo;
        const float val = acc[mi][ni][r] + bia[c];
        const int b = m >> 11, q = m & (N_ - 1);
        const int h = c >> 6,  d = c & (HD_ - 1);
        const size_t bh = (size_t)b * H_ + h;
        if (mode == 0)
          Qh[(bh * N_ + q) * HD_ + d] = (_Float16)(val * 0.125f);  // 1/sqrt(64)
        else if (mode == 1)
          Kh[(bh * N_ + q) * HD_ + d] = (_Float16)val;
        else
          Vt[(bh * HD_ + d) * N_ + q] = (_Float16)val;  // transposed for PV
      }
}

// ---------------------------------------------------------------------------
// Flash attention. 128 threads = 4 waves; block handles 64 q-rows of one
// (b,h). Wave w owns q-rows [w*16, w*16+16). Loop over 64-key blocks:
//   prefetch next K/V block -> S-frags (WMMA, K streamed from global [n][d])
//   -> S_lds (+ mask bias) -> per-row online softmax -> rescale O -> O += P*V
//   (WMMA, V^T streamed from global [d][n]).
// ---------------------------------------------------------------------------
__global__ __launch_bounds__(128) void attn_kernel(
    const _Float16* __restrict__ Qh,   // [BH, N, 64] pre-scaled
    const _Float16* __restrict__ Kh,   // [BH, N, 64]
    const _Float16* __restrict__ Vt,   // [BH, 64, N]
    const float* __restrict__ mask,    // [B, N]
    _Float16* __restrict__ ctx) {      // [B, N, H*64] merged heads
  const int bh = blockIdx.y;           // 0..31
  const int b = bh / H_, h = bh % H_;
  const int qt = blockIdx.x;           // 64-row q tile, 0..31
  const int tid = threadIdx.x;
  const int w = tid >> 5, lane = tid & 31, lo = lane & 15, hi = lane >> 4;

  __shared__ __align__(16) float    S_lds[64][68];
  __shared__ __align__(16) _Float16 P_lds[64][72];
  __shared__ float mrow[64], lrow[64], arow[64];

  // Q fragments for this wave's 16 rows (two 32-wide k-chunks over d=64)
  const _Float16* Qb = Qh + ((size_t)bh * N_ + qt * 64 + w * 16) * HD_;
  v16h qf[2];
#pragma unroll
  for (int kc = 0; kc < 2; ++kc) qf[kc] = load_frag(Qb + kc * 32, HD_, lo, hi);

  v8f o[4] = {};
  if (tid < 64) { mrow[tid] = -1e30f; lrow[tid] = 0.0f; }
  __syncthreads();

  const float* maskb = mask + (size_t)b * N_;
  const _Float16* Kbase = Kh + (size_t)bh * N_ * HD_;
  const _Float16* Vbase = Vt + (size_t)bh * HD_ * N_;

  for (int kb = 0; kb < N_; kb += 64) {
    // prefetch next key block (K: 8KB contiguous; V: 64 rows x 128B)
    if (kb + 64 < N_) {
      __builtin_prefetch(Kbase + (size_t)(kb + 64) * HD_ + tid * 32, 0, 0);
      __builtin_prefetch(Vbase + (size_t)(tid & 63) * N_ + kb + 64, 0, 0);
    }

    // ---- S = Q K^T (per wave: 16 q-rows x 64 key-cols) ----
    v8f s[4];
    float mb[4];
#pragma unroll
    for (int nf = 0; nf < 4; ++nf) {
      mb[nf] = -1e6f * (1.0f - maskb[kb + nf * 16 + lo]);
      v8f a = {};
      const _Float16* Kb = Kbase + (size_t)(kb + nf * 16) * HD_;
#pragma unroll
      for (int kc = 0; kc < 2; ++kc)
        a = wmma_f16(qf[kc], load_frag(Kb + kc * 32, HD_, lo, hi), a);
      s[nf] = a;
    }
    __syncthreads();  // prior iter's S_lds / P_lds readers are done
#pragma unroll
    for (int nf = 0; nf < 4; ++nf)
#pragma unroll
      for (int r = 0; r < 8; ++r)
        S_lds[w * 16 + hi * 8 + r][nf * 16 + lo] = s[nf][r] + mb[nf];
    __syncthreads();

    // ---- online softmax, one thread per q-row ----
    if (tid < 64) {
      const int row = tid;
      float m_old = mrow[row];
      float mx = m_old;
#pragma unroll 4
      for (int c = 0; c < 64; ++c) mx = fmaxf(mx, S_lds[row][c]);
      const float alpha = __expf(m_old - mx);
      float l = alpha * lrow[row];
#pragma unroll 4
      for (int c = 0; c < 64; ++c) {
        const float p = __expf(S_lds[row][c] - mx);
        l += p;
        P_lds[row][c] = (_Float16)p;
      }
      mrow[row] = mx; lrow[row] = l; arow[row] = alpha;
    }
    __syncthreads();

    // ---- rescale O, then O += P V ----
    float a8[8];
#pragma unroll
    for (int r = 0; r < 8; ++r) a8[r] = arow[w * 16 + hi * 8 + r];
#pragma unroll
    for (int nf = 0; nf < 4; ++nf)
#pragma unroll
      for (int r = 0; r < 8; ++r) o[nf][r] *= a8[r];

    v16h pf[2];
#pragma unroll
    for (int kc = 0; kc < 2; ++kc)
      pf[kc] = load_frag(&P_lds[w * 16][kc * 32], 72, lo, hi);
#pragma unroll
    for (int nf = 0; nf < 4; ++nf) {
      const _Float16* Vb = Vbase + (size_t)(nf * 16) * N_ + kb;
#pragma unroll
      for (int kc = 0; kc < 2; ++kc)
        o[nf] = wmma_f16(pf[kc], load_frag(Vb + kc * 32, N_, lo, hi), o[nf]);
    }
  }

  // ---- finalize: divide by l, write merged-head ctx in f16 ----
  __syncthreads();
  if (tid < 64) lrow[tid] = 1.0f / fmaxf(lrow[tid], 1e-20f);
  __syncthreads();
  float inv8[8];
#pragma unroll
  for (int r = 0; r < 8; ++r) inv8[r] = lrow[w * 16 + hi * 8 + r];
#pragma unroll
  for (int nf = 0; nf < 4; ++nf)
#pragma unroll
    for (int r = 0; r < 8; ++r) {
      const int q = qt * 64 + w * 16 + hi * 8 + r;
      const int d = nf * 16 + lo;
      ctx[((size_t)b * N_ + q) * D_ + h * HD_ + d] =
          (_Float16)(o[nf][r] * inv8[r]);
    }
}

// ---------------------------------------------------------------------------
// Output projection: out = ctx @ Wo + bo, fp32 output. Same async tiling.
// ---------------------------------------------------------------------------
__global__ __launch_bounds__(256) void oproj_gemm(
    const _Float16* __restrict__ Ah,   // ctx [4096, 1024] f16
    const _Float16* __restrict__ WT,   // Wo^T [1024(N)][1024(K)] f16
    const float* __restrict__ bo,
    float* __restrict__ out) {
  const int Mb = blockIdx.y * 128;
  const int Nb = blockIdx.x * 64;
  const int tid = threadIdx.x;
  const int w = tid >> 5, lane = tid & 31, lo = lane & 15, hi = lane >> 4;
  const int wm = w & 3, wn = w >> 2;

  __shared__ __align__(16) _Float16 As[128][40];
  __shared__ __align__(16) _Float16 Bs[64][40];

  v8f acc[2][2] = {};

  const int ar = tid >> 1;
  const int ac = (tid & 1) * 16;
  const int bn = tid >> 2;
  const int bk8 = (tid & 3) * 8;
  const unsigned a_dst = lds_addr(&As[ar][ac]);
  const unsigned b_dst = lds_addr(&Bs[bn][bk8]);

  for (int k0 = 0; k0 < D_; k0 += 32) {
    __syncthreads();
    {
      const _Float16* srcA = Ah + (size_t)(Mb + ar) * D_ + k0 + ac;
      async_g2l_b128(a_dst,      srcA);
      async_g2l_b128(a_dst + 16, srcA + 8);
      const _Float16* srcB = WT + (size_t)(Nb + bn) * D_ + k0 + bk8;
      async_g2l_b128(b_dst, srcB);
    }
    wait_async0();
    __syncthreads();

    v16h af[2], bf[2];
#pragma unroll
    for (int mi = 0; mi < 2; ++mi)
      af[mi] = load_frag(&As[wm * 32 + mi * 16][0], 40, lo, hi);
#pragma unroll
    for (int ni = 0; ni < 2; ++ni)
      bf[ni] = load_frag(&Bs[wn * 32 + ni * 16][0], 40, lo, hi);
#pragma unroll
    for (int mi = 0; mi < 2; ++mi)
#pragma unroll
      for (int ni = 0; ni < 2; ++ni)
        acc[mi][ni] = wmma_f16(af[mi], bf[ni], acc[mi][ni]);
  }

#pragma unroll
  for (int mi = 0; mi < 2; ++mi)
#pragma unroll
    for (int ni = 0; ni < 2; ++ni)
#pragma unroll
      for (int r = 0; r < 8; ++r) {
        const int m = Mb + wm * 32 + mi * 16 + hi * 8 + r;
        const int c = Nb + wn * 32 + ni * 16 + lo;
        out[(size_t)m * D_ + c] = acc[mi][ni][r] + bo[c];
      }
}

// ---------------------------------------------------------------------------
// Host side
// ---------------------------------------------------------------------------
extern "C" void kernel_launch(void* const* d_in, const int* in_sizes, int n_in,
                              void* d_out, int out_size, void* d_ws,
                              size_t ws_size, hipStream_t stream) {
  const float* X    = (const float*)d_in[0];
  const float* mask = (const float*)d_in[1];
  const float* Wq   = (const float*)d_in[2];
  const float* bq   = (const float*)d_in[3];
  const float* Wk   = (const float*)d_in[4];
  const float* bk   = (const float*)d_in[5];
  const float* Wv   = (const float*)d_in[6];
  const float* bv   = (const float*)d_in[7];
  const float* Wo   = (const float*)d_in[8];
  const float* bo   = (const float*)d_in[9];
  float* out = (float*)d_out;

  // scratch layout (f16 buffers), ~48 MB total
  char* ws = (char*)d_ws;
  _Float16* Xh  = (_Float16*)ws;  ws += (size_t)M_ * D_ * 2;        // 8 MB
  _Float16* WqT = (_Float16*)ws;  ws += (size_t)D_ * D_ * 2;        // 2 MB
  _Float16* WkT = (_Float16*)ws;  ws += (size_t)D_ * D_ * 2;
  _Float16* WvT = (_Float16*)ws;  ws += (size_t)D_ * D_ * 2;
  _Float16* WoT = (_Float16*)ws;  ws += (size_t)D_ * D_ * 2;
  _Float16* Qh  = (_Float16*)ws;  ws += (size_t)M_ * D_ * 2;        // 8 MB
  _Float16* Kh  = (_Float16*)ws;  ws += (size_t)M_ * D_ * 2;
  _Float16* Vt  = (_Float16*)ws;  ws += (size_t)M_ * D_ * 2;
  _Float16* Ctx = (_Float16*)ws;  ws += (size_t)M_ * D_ * 2;

  f32_to_f16<<<1024, 256, 0, stream>>>(X, Xh, M_ * D_);
  convT_f16<<<dim3(32, 32), 256, 0, stream>>>(Wq, WqT);
  convT_f16<<<dim3(32, 32), 256, 0, stream>>>(Wk, WkT);
  convT_f16<<<dim3(32, 32), 256, 0, stream>>>(Wv, WvT);
  convT_f16<<<dim3(32, 32), 256, 0, stream>>>(Wo, WoT);

  // QKV projections: grid (Ntiles=16, Mtiles=32, 3 matrices)
  qkv_gemm<<<dim3(16, 32, 3), 256, 0, stream>>>(Xh, WqT, WkT, WvT, bq, bk, bv,
                                                Qh, Kh, Vt);

  // attention: grid (32 q-tiles, 32 bh)
  attn_kernel<<<dim3(32, 32), 128, 0, stream>>>(Qh, Kh, Vt, mask, Ctx);

  // output projection
  oproj_gemm<<<dim3(16, 32), 256, 0, stream>>>(Ctx, WoT, bo, out);
}
